// Model_9534827397374
// MI455X (gfx1250) — matrix operand order repaired and checked by reading, not compiled, not run on hardware
//
#include <hip/hip_runtime.h>
#include <hip/hip_bf16.h>
#include <math.h>

// ---------------------------------------------------------------------------
// Types for CDNA5 WMMA
// ---------------------------------------------------------------------------
typedef __bf16 bf16_t;
typedef __attribute__((ext_vector_type(16))) __bf16 v16bf;
typedef __attribute__((ext_vector_type(8)))  __bf16 v8bf;
typedef __attribute__((ext_vector_type(4)))  __bf16 v4bf;
typedef __attribute__((ext_vector_type(8)))  float  v8f;
typedef __attribute__((ext_vector_type(4)))  float  v4f;
typedef __attribute__((ext_vector_type(4)))  int    v4i;

// ---------------------------------------------------------------------------
// CDNA5 async global->LDS staging (ASYNCcnt), with portable fallback.
// Builtin signature (from probe): (as1 int4* gaddr, as3 int4* lds, Ii off, Ii cpol)
// ---------------------------------------------------------------------------
#if defined(__has_builtin)
#  if __has_builtin(__builtin_amdgcn_global_load_async_to_lds_b128)
#    define USE_ASYNC_LDS 1
#  endif
#  if __has_builtin(__builtin_amdgcn_s_wait_asynccnt)
#    define HAVE_WAIT_ASYNC 1
#  endif
#endif

__device__ __forceinline__ void copy16_g2l(const bf16_t* __restrict__ g, bf16_t* l)
{
#if defined(USE_ASYNC_LDS)
    __builtin_amdgcn_global_load_async_to_lds_b128(
        (__attribute__((address_space(1))) v4i*)g,
        (__attribute__((address_space(3))) v4i*)l, 0, 0);
#else
    *(v8bf*)l = *(const v8bf*)g;
#endif
}

__device__ __forceinline__ void wait_async_all()
{
#if defined(USE_ASYNC_LDS)
#  if defined(HAVE_WAIT_ASYNC)
    __builtin_amdgcn_s_wait_asynccnt(0);
#  else
    asm volatile("s_wait_asynccnt 0" ::: "memory");
#  endif
#endif
}

// ---------------------------------------------------------------------------
// bf16 WMMA GEMM:  Out[M,N] = act( alpha*(A@B) + colAdd[n] + rowAdd[m] )
//   A: bf16 [M,K] row-major (lda);  B: bf16 [N,K] row-major (ldb)  (both K-major)
//   Out: OUTBF ? bf16 : f32, row-major (ldo)
// Tile 128x128, K-step 32; 8 waves (4M x 2N), wave tile 32x64 = 2x4 WMMA accs.
// Staging: 16B/lane async global->LDS copies (2 per thread per matrix per step).
// ---------------------------------------------------------------------------
#define GBM 128
#define GBN 128
#define GBK 32

template<bool RELU, bool HAS_COL, bool HAS_ROW, bool OUTBF>
__global__ __launch_bounds__(256)
void gemm_bf16_wmma(const bf16_t* __restrict__ A, const bf16_t* __restrict__ Bm,
                    void* __restrict__ OutP,
                    int M, int N, int K, int lda, int ldb, int ldo,
                    float alpha,
                    const float* __restrict__ colAdd,
                    const float* __restrict__ rowAdd)
{
    __shared__ __align__(64) bf16_t Asb[GBM * GBK];   // [m][k]
    __shared__ __align__(64) bf16_t Bsb[GBN * GBK];   // [n][k]

    const int tid  = threadIdx.x;
    const int wave = tid >> 5;
    const int lane = tid & 31;
    const int wm   = wave & 3;
    const int wn   = wave >> 2;
    const int tileM = blockIdx.x * GBM;
    const int tileN = blockIdx.y * GBN;
    const bool fullM = (tileM + GBM) <= M;
    const bool fullN = (tileN + GBN) <= N;

    v8f acc[2][4] = {};

    for (int k0 = 0; k0 < K; k0 += GBK) {
        // ---- stage A tile: 512 groups of 16B (2 per thread) ----
        #pragma unroll
        for (int i = 0; i < 2; ++i) {
            int g  = tid + i * 256;
            int m  = g >> 2;
            int kc = (g & 3) << 3;
            int gm = tileM + m; if (!fullM && gm >= M) gm = M - 1;
            copy16_g2l(A + (long long)gm * lda + (k0 + kc), &Asb[m * GBK + kc]);
        }
        // ---- stage B tile ----
        #pragma unroll
        for (int i = 0; i < 2; ++i) {
            int g  = tid + i * 256;
            int n  = g >> 2;
            int kc = (g & 3) << 3;
            int gn = tileN + n; if (!fullN && gn >= N) gn = N - 1;
            copy16_g2l(Bm + (long long)gn * ldb + (k0 + kc), &Bsb[n * GBK + kc]);
        }
        wait_async_all();
        __syncthreads();

        // ---- fragments + WMMA (wave tile 32x64 = 2x4 subtiles) ----
        {
            const int lrow = lane & 15;
            const int ah   = (lane >> 4) << 4;   // A K-half offset: 0 or 16
            const int bh   = (lane >> 4) << 4;   // B K-half offset: 0 or 16
            v16bf bfrag[4];
            #pragma unroll
            for (int ni = 0; ni < 4; ++ni)
                bfrag[ni] = *(const v16bf*)(&Bsb[(wn * 64 + ni * 16 + lrow) * GBK] + bh);
            #pragma unroll
            for (int mi = 0; mi < 2; ++mi) {
                const bf16_t* ap = &Asb[(wm * 32 + mi * 16 + lrow) * GBK];
                v8bf a_lo = *(const v8bf*)(ap + ah);
                v8bf a_hi = *(const v8bf*)(ap + ah + 8);
                v16bf afrag = __builtin_shufflevector(a_lo, a_hi,
                    0,1,2,3,4,5,6,7,8,9,10,11,12,13,14,15);
                #pragma unroll
                for (int ni = 0; ni < 4; ++ni)
                    acc[mi][ni] = __builtin_amdgcn_wmma_f32_16x16x32_bf16(
                        false, afrag, false, bfrag[ni], (short)0, acc[mi][ni], false, false);
            }
        }
        __syncthreads();
    }

    // ---- epilogue ----
    float*  Outf = (float*)OutP;
    bf16_t* Outb = (bf16_t*)OutP;
    const int lrow = lane & 15;
    const int rowh = (lane >> 4) << 3;
    if (fullM && fullN) {
        #pragma unroll
        for (int ni = 0; ni < 4; ++ni) {
            int n = tileN + wn * 64 + ni * 16 + lrow;
            float cadd = HAS_COL ? colAdd[n] : 0.0f;
            #pragma unroll
            for (int mi = 0; mi < 2; ++mi) {
                int mbase = tileM + wm * 32 + mi * 16 + rowh;
                #pragma unroll
                for (int r = 0; r < 8; ++r) {
                    float v = alpha * acc[mi][ni][r] + cadd;
                    if (HAS_ROW) v += rowAdd[mbase + r];
                    if (RELU) v = fmaxf(v, 0.0f);
                    if (OUTBF) Outb[(long long)(mbase + r) * ldo + n] = (bf16_t)v;
                    else       Outf[(long long)(mbase + r) * ldo + n] = v;
                }
            }
        }
    } else {
        #pragma unroll
        for (int ni = 0; ni < 4; ++ni) {
            int n  = tileN + wn * 64 + ni * 16 + lrow;
            int nc = (n < N) ? n : (N - 1);
            float cadd = HAS_COL ? colAdd[nc] : 0.0f;
            #pragma unroll
            for (int mi = 0; mi < 2; ++mi) {
                int mbase = tileM + wm * 32 + mi * 16 + rowh;
                #pragma unroll
                for (int r = 0; r < 8; ++r) {
                    int m = mbase + r;
                    if (m < M && n < N) {
                        float v = alpha * acc[mi][ni][r] + cadd;
                        if (HAS_ROW) v += rowAdd[m];
                        if (RELU) v = fmaxf(v, 0.0f);
                        if (OUTBF) Outb[(long long)m * ldo + n] = (bf16_t)v;
                        else       Outf[(long long)m * ldo + n] = v;
                    }
                }
            }
        }
    }
}

// ---------------------------------------------------------------------------
// f32 -> bf16 conversion (vectorized x4; n4 = count/4, count % 4 == 0)
// ---------------------------------------------------------------------------
__global__ __launch_bounds__(256)
void f32_to_bf16_kernel(const float* __restrict__ src, bf16_t* __restrict__ dst,
                        long long n4)
{
    long long i = (long long)blockIdx.x * 256 + threadIdx.x;
    if (i >= n4) return;
    v4f v = *(const v4f*)(src + i * 4);
    v4bf h; h[0]=(bf16_t)v[0]; h[1]=(bf16_t)v[1]; h[2]=(bf16_t)v[2]; h[3]=(bf16_t)v[3];
    *(v4bf*)(dst + i * 4) = h;
}

// ---------------------------------------------------------------------------
// Transpose + convert weights:  Wt[n*K + k] = (bf16)W[k*N + n].  grid = N.
// ---------------------------------------------------------------------------
__global__ __launch_bounds__(256)
void transpose_bf16_kernel(const float* __restrict__ W, bf16_t* __restrict__ Wt,
                           int K, int N)
{
    int n = blockIdx.x;
    for (int k = threadIdx.x; k < K; k += 256)
        Wt[(long long)n * K + k] = (bf16_t)W[(long long)k * N + n];
}

// ---------------------------------------------------------------------------
// Row squared norms over bf16 rows: out[m] = sum_d X[m,d]^2.  One wave32/row.
// ---------------------------------------------------------------------------
__global__ __launch_bounds__(256)
void rownorm_bf_kernel(const bf16_t* __restrict__ X, int M, int D,
                       float* __restrict__ out)
{
    int row  = blockIdx.x * 8 + (threadIdx.x >> 5);
    int lane = threadIdx.x & 31;
    if (row >= M) return;                       // uniform per wave
    float s = 0.0f;
    for (int d = lane; d < D; d += 32) { float v = (float)X[(long long)row * D + d]; s += v * v; }
    #pragma unroll
    for (int o = 16; o > 0; o >>= 1) s += __shfl_xor(s, o, 32);
    if (lane == 0) out[row] = s;
}

// ---------------------------------------------------------------------------
// Exact top-C (smallest) per row: 4-pass 8-bit radix select + deterministic
// scan-based collection.  One 256-thread workgroup per row.
// ---------------------------------------------------------------------------
__device__ __forceinline__ unsigned fkey(float f) {
    unsigned u = __float_as_uint(f);
    return (u & 0x80000000u) ? ~u : (u | 0x80000000u);   // monotonic ascending
}

__global__ __launch_bounds__(256)
void topk_select_kernel(const float* __restrict__ d2, int N, int C,
                        int rowOffset, int* __restrict__ I)
{
    const int rb  = blockIdx.x;
    const int tid = threadIdx.x;
    const float* row = d2 + (long long)rb * N;

    __shared__ unsigned hist[256];
    __shared__ unsigned s_prefix;
    __shared__ int      s_want;
    __shared__ int      scanL[256];
    __shared__ int      scanE[256];
    __shared__ int      s_totLess;

    if (tid == 0) { s_prefix = 0u; s_want = C; }
    __syncthreads();

    for (int shift = 24; shift >= 0; shift -= 8) {
        hist[tid] = 0u;
        __syncthreads();
        unsigned pfx = s_prefix;
        for (int j = tid; j < N; j += 256) {
            unsigned key = fkey(row[j]);
            bool match = (shift == 24) || ((key >> (shift + 8)) == (pfx >> (shift + 8)));
            if (match) atomicAdd(&hist[(key >> shift) & 255u], 1u);
        }
        __syncthreads();
        if (tid == 0) {
            unsigned cum = 0; int want = s_want; unsigned b = 0;
            for (b = 0; b < 256u; ++b) {
                unsigned c = hist[b];
                if (cum + c >= (unsigned)want) break;
                cum += c;
            }
            s_prefix = pfx | (b << shift);
            s_want   = want - (int)cum;
        }
        __syncthreads();
    }
    const unsigned T = s_prefix;   // key of the C-th smallest element

    int myless = 0, myeq = 0;
    for (int j = tid; j < N; j += 256) {
        unsigned key = fkey(row[j]);
        if (key < T) myless++; else if (key == T) myeq++;
    }
    scanL[tid] = myless; scanE[tid] = myeq;
    __syncthreads();
    if (tid == 0) {
        int aL = 0, aE = 0;
        for (int t = 0; t < 256; ++t) {
            int l = scanL[t], e = scanE[t];
            scanL[t] = aL; scanE[t] = aE;
            aL += l; aE += e;
        }
        s_totLess = aL;
    }
    __syncthreads();

    int lpos = scanL[tid];
    int epos = s_totLess + scanE[tid];
    int* Irow = I + (long long)(rowOffset + rb) * C;
    for (int j = tid; j < N; j += 256) {
        unsigned key = fkey(row[j]);
        if (key < T)       { Irow[lpos++] = j; }
        else if (key == T) { if (epos < C) Irow[epos] = j; epos++; }
    }
}

// ---------------------------------------------------------------------------
// Gather S = d2[b, I], softmax over C; wy[b] = sum_c w_c * cand_y[I_c].
// ---------------------------------------------------------------------------
__global__ __launch_bounds__(128)
void softmax_gather_kernel(const float* __restrict__ d2chunk, int N, int C,
                           int rowOffset, const int* __restrict__ I,
                           const float* __restrict__ cand_y,
                           float* __restrict__ weights, float* __restrict__ wy)
{
    const int rb  = blockIdx.x;
    const int row = rowOffset + rb;
    const int tid = threadIdx.x;
    __shared__ float red[128];

    float s = -INFINITY, y = 0.0f;
    if (tid < C) {
        int idx = I[(long long)row * C + tid];
        s = d2chunk[(long long)rb * N + idx];
        y = cand_y[idx];
    }
    red[tid] = s; __syncthreads();
    for (int o = 64; o > 0; o >>= 1) { if (tid < o) red[tid] = fmaxf(red[tid], red[tid + o]); __syncthreads(); }
    float mx = red[0]; __syncthreads();
    float e = (tid < C) ? __expf(s - mx) : 0.0f;
    red[tid] = e; __syncthreads();
    for (int o = 64; o > 0; o >>= 1) { if (tid < o) red[tid] += red[tid + o]; __syncthreads(); }
    float inv = 1.0f / red[0]; __syncthreads();
    float w = e * inv;
    if (tid < C) weights[(long long)row * C + tid] = w;
    red[tid] = w * y; __syncthreads();
    for (int o = 64; o > 0; o >>= 1) { if (tid < o) red[tid] += red[tid + o]; __syncthreads(); }
    if (tid == 0) wy[row] = red[0];
}

// ---------------------------------------------------------------------------
// Dif_bf[rb*C+c, d] = (bf16)( k_bf[row,d] - ki_bf[I[row,c], d] )
// ---------------------------------------------------------------------------
__global__ __launch_bounds__(256)
void build_diff_kernel(const bf16_t* __restrict__ kmat_bf, const bf16_t* __restrict__ ki_bf,
                       const int* __restrict__ I, int C, int D, int rowOffset,
                       bf16_t* __restrict__ Dif)
{
    int cr  = blockIdx.x;
    int rb  = cr / C, c = cr % C;
    int row = rowOffset + rb;
    int idx = I[(long long)row * C + c];
    int d   = threadIdx.x;
    Dif[(long long)cr * D + d] =
        (bf16_t)((float)kmat_bf[(long long)row * D + d] - (float)ki_bf[(long long)idx * D + d]);
}

// ---------------------------------------------------------------------------
// x[row,d] = xenc[row,d] + b_Y[d] + wy[row]*W_Y[d] + sum_c w_c * T2[rb*C+c, d]
// ---------------------------------------------------------------------------
__global__ __launch_bounds__(256)
void context_reduce_kernel(const float* __restrict__ T2, const float* __restrict__ weights,
                           const float* __restrict__ wy, const float* __restrict__ xenc,
                           const float* __restrict__ W_Y, const float* __restrict__ b_Y,
                           int C, int D, int rowOffset, float* __restrict__ x)
{
    int rb  = blockIdx.x;
    int row = rowOffset + rb;
    int d   = threadIdx.x;
    const float* wrow = weights + (long long)row * C;
    float acc = 0.0f;
    for (int c = 0; c < C; ++c) acc += wrow[c] * T2[(long long)(rb * C + c) * D + d];
    x[(long long)row * D + d] =
        xenc[(long long)row * D + d] + b_Y[d] + wy[row] * W_Y[d] + acc;
}

// ---------------------------------------------------------------------------
// Tail: LN -> Linear(256,512) -> ReLU -> Linear(512,256) (+res) -> LN -> ReLU
// -> dot W_out.  One 256-thread workgroup per row.
// ---------------------------------------------------------------------------
__global__ __launch_bounds__(256)
void head_kernel(const float* __restrict__ x,
                 const float* __restrict__ lnpg, const float* __restrict__ lnpb,
                 const float* __restrict__ W_P1, const float* __restrict__ b_P1,
                 const float* __restrict__ W_P2, const float* __restrict__ b_P2,
                 const float* __restrict__ lnog, const float* __restrict__ lnob,
                 const float* __restrict__ W_out, const float* __restrict__ b_out,
                 float* __restrict__ out)
{
    const int row = blockIdx.x, tid = threadIdx.x;
    __shared__ float xs[256];
    __shared__ float hbuf[512];
    __shared__ float red[256];

    float xv = x[(long long)row * 256 + tid];

    red[tid] = xv; __syncthreads();
    for (int o = 128; o > 0; o >>= 1) { if (tid < o) red[tid] += red[tid + o]; __syncthreads(); }
    float m = red[0] * (1.0f / 256.0f); __syncthreads();
    float dv = xv - m;
    red[tid] = dv * dv; __syncthreads();
    for (int o = 128; o > 0; o >>= 1) { if (tid < o) red[tid] += red[tid + o]; __syncthreads(); }
    float var = red[0] * (1.0f / 256.0f); __syncthreads();
    xs[tid] = dv * rsqrtf(var + 1e-5f) * lnpg[tid] + lnpb[tid];
    __syncthreads();

    for (int j = tid; j < 512; j += 256) {
        float o1 = b_P1[j];
        for (int d = 0; d < 256; ++d) o1 += xs[d] * W_P1[d * 512 + j];
        hbuf[j] = fmaxf(o1, 0.0f);
    }
    __syncthreads();

    float o2 = b_P2[tid];
    for (int j = 0; j < 512; ++j) o2 += hbuf[j] * W_P2[j * 256 + tid];
    float x2 = xv + o2;

    red[tid] = x2; __syncthreads();
    for (int o = 128; o > 0; o >>= 1) { if (tid < o) red[tid] += red[tid + o]; __syncthreads(); }
    float m2 = red[0] * (1.0f / 256.0f); __syncthreads();
    float dv2 = x2 - m2;
    red[tid] = dv2 * dv2; __syncthreads();
    for (int o = 128; o > 0; o >>= 1) { if (tid < o) red[tid] += red[tid + o]; __syncthreads(); }
    float var2 = red[0] * (1.0f / 256.0f); __syncthreads();
    float r = fmaxf(dv2 * rsqrtf(var2 + 1e-5f) * lnog[tid] + lnob[tid], 0.0f);
    red[tid] = r * W_out[tid]; __syncthreads();
    for (int o = 128; o > 0; o >>= 1) { if (tid < o) red[tid] += red[tid + o]; __syncthreads(); }
    if (tid == 0) out[row] = red[0] + b_out[0];
}

// ---------------------------------------------------------------------------
// Host-side orchestration.  All GEMM operands live in bf16 (K-major both
// sides); d2 chunked (256 rows x 100k f32 = 102 MB) so the distance tile +
// ki_bf (51 MB) stay ~L2-resident (192 MB).  Workspace ~59M floats (~236 MB).
// ---------------------------------------------------------------------------
extern "C" void kernel_launch(void* const* d_in, const int* in_sizes, int n_in,
                              void* d_out, int out_size, void* d_ws, size_t ws_size,
                              hipStream_t stream)
{
    (void)in_sizes; (void)n_in; (void)out_size; (void)ws_size;

    const float* x_num    = (const float*)d_in[0];
    const float* cand_num = (const float*)d_in[1];
    const float* cand_y   = (const float*)d_in[2];
    const float* W_lin    = (const float*)d_in[3];
    const float* b_lin    = (const float*)d_in[4];
    const float* W_K      = (const float*)d_in[5];
    const float* b_K      = (const float*)d_in[6];
    const float* W_Y      = (const float*)d_in[7];
    const float* b_Y      = (const float*)d_in[8];
    const float* W_T1     = (const float*)d_in[9];
    const float* b_T1     = (const float*)d_in[10];
    const float* W_T2     = (const float*)d_in[11];
    const float* ln_p_g   = (const float*)d_in[12];
    const float* ln_p_b   = (const float*)d_in[13];
    const float* W_P1     = (const float*)d_in[14];
    const float* b_P1     = (const float*)d_in[15];
    const float* W_P2     = (const float*)d_in[16];
    const float* b_P2     = (const float*)d_in[17];
    const float* ln_o_g   = (const float*)d_in[18];
    const float* ln_o_b   = (const float*)d_in[19];
    const float* W_out    = (const float*)d_in[20];
    const float* b_out    = (const float*)d_in[21];

    constexpr int B = 1024, N = 100000, F = 128, D = 256, DI = 512, C = 96;
    constexpr int BCHUNK = 256;
    constexpr int CR = BCHUNK * C;                // 24576 context rows per chunk

    // ---- workspace carve-up (float units; all offsets 16B-aligned) ----
    float* ws = (float*)d_ws;
    long long off = 0;
    auto takeF = [&](long long n) { float* p = ws + off; off += (n + 63) & ~63LL; return p; };
    float*  scr      = takeF(25600000);                     // d2 chunk / (Dif|H|T2)
    bf16_t* ki_bf    = (bf16_t*)takeF((long long)N * D / 2);      // 12.8M fl
    bf16_t* cnum_bf  = (bf16_t*)takeF((long long)N * F / 2);      //  6.4M fl
    bf16_t* htmp_bf  = (bf16_t*)takeF((long long)N * D / 2);      // 12.8M fl
    float*  xenc     = takeF((long long)B * D);
    float*  xbuf     = takeF((long long)B * D);
    bf16_t* xnum_bf  = (bf16_t*)takeF((long long)B * F / 2);
    bf16_t* xenc_bf  = (bf16_t*)takeF((long long)B * D / 2);
    bf16_t* kmat_bf  = (bf16_t*)takeF((long long)B * D / 2);
    bf16_t* WlinT    = (bf16_t*)takeF((long long)D * F / 2);
    bf16_t* WKT      = (bf16_t*)takeF((long long)D * D / 2);
    bf16_t* WT1T     = (bf16_t*)takeF((long long)DI * D / 2);
    bf16_t* WT2T     = (bf16_t*)takeF((long long)D * DI / 2);
    float*  knorm    = takeF(B);
    float*  kinorm   = takeF(N);
    float*  weights  = takeF((long long)B * C);
    float*  wy       = takeF(B);
    int*    I        = (int*)takeF((long long)B * C);

    dim3 blk(256);

    // ---- one-time conversions / weight transposes (tiny) ----
    f32_to_bf16_kernel<<<dim3((N * F / 4 + 255) / 256), blk, 0, stream>>>(cand_num, cnum_bf, (long long)N * F / 4);
    f32_to_bf16_kernel<<<dim3((B * F / 4 + 255) / 256), blk, 0, stream>>>(x_num, xnum_bf, (long long)B * F / 4);
    transpose_bf16_kernel<<<dim3(D),  blk, 0, stream>>>(W_lin, WlinT, F, D);   // [F,D] -> [D,F]
    transpose_bf16_kernel<<<dim3(D),  blk, 0, stream>>>(W_K,   WKT,   D, D);   // [D,D] -> [D,D]
    transpose_bf16_kernel<<<dim3(DI), blk, 0, stream>>>(W_T1,  WT1T,  D, DI);  // [D,DI] -> [DI,D]
    transpose_bf16_kernel<<<dim3(D),  blk, 0, stream>>>(W_T2,  WT2T,  DI, D);  // [DI,D] -> [D,DI]

    // ---- encoder (bf16 operands, bf16 outputs where only GEMMs consume) ----
    gemm_bf16_wmma<false, true, false, true>
        <<<dim3((N + GBM - 1) / GBM, D / GBN), blk, 0, stream>>>(
        cnum_bf, WlinT, htmp_bf, N, D, F, F, F, D, 1.0f, b_lin, nullptr);
    gemm_bf16_wmma<false, true, false, true>
        <<<dim3((N + GBM - 1) / GBM, D / GBN), blk, 0, stream>>>(
        htmp_bf, WKT, ki_bf, N, D, D, D, D, D, 1.0f, b_K, nullptr);
    gemm_bf16_wmma<false, true, false, false>
        <<<dim3(B / GBM, D / GBN), blk, 0, stream>>>(
        xnum_bf, WlinT, xenc, B, D, F, F, F, D, 1.0f, b_lin, nullptr);
    f32_to_bf16_kernel<<<dim3((B * D / 4 + 255) / 256), blk, 0, stream>>>(xenc, xenc_bf, (long long)B * D / 4);
    gemm_bf16_wmma<false, true, false, true>
        <<<dim3(B / GBM, D / GBN), blk, 0, stream>>>(
        xenc_bf, WKT, kmat_bf, B, D, D, D, D, D, 1.0f, b_K, nullptr);

    // ---- squared norms (bf16-consistent with the WMMA dot products) ----
    rownorm_bf_kernel<<<dim3((N + 7) / 8), blk, 0, stream>>>(ki_bf, N, D, kinorm);
    rownorm_bf_kernel<<<dim3((B + 7) / 8), blk, 0, stream>>>(kmat_bf, B, D, knorm);

    // ---- distances + exact top-96 + softmax, chunked over batch rows ----
    for (int c0 = 0; c0 < B; c0 += BCHUNK) {
        gemm_bf16_wmma<false, true, true, false>
            <<<dim3(BCHUNK / GBM, (N + GBN - 1) / GBN), blk, 0, stream>>>(
            kmat_bf + (long long)c0 * D, ki_bf, scr,
            BCHUNK, N, D, D, D, N, -2.0f, kinorm, knorm + c0);
        topk_select_kernel<<<dim3(BCHUNK), blk, 0, stream>>>(scr, N, C, c0, I);
        softmax_gather_kernel<<<dim3(BCHUNK), dim3(128), 0, stream>>>(
            scr, N, C, c0, I, cand_y, weights, wy);
    }

    // ---- context T-MLP + weighted reduction (scr reused: Dif_bf|H_bf|T2) ----
    bf16_t* Dif = (bf16_t*)scr;                               // CR*D bf16
    bf16_t* H   = (bf16_t*)(scr + (long long)CR * D / 2);     // CR*DI bf16
    float*  T2  = scr + (long long)CR * D / 2 + (long long)CR * DI / 2;  // CR*D f32
    for (int c0 = 0; c0 < B; c0 += BCHUNK) {
        build_diff_kernel<<<dim3(CR), blk, 0, stream>>>(kmat_bf, ki_bf, I, C, D, c0, Dif);
        gemm_bf16_wmma<true, true, false, true>
            <<<dim3(CR / GBM, DI / GBN), blk, 0, stream>>>(
            Dif, WT1T, H, CR, DI, D, D, D, DI, 1.0f, b_T1, nullptr);
        gemm_bf16_wmma<false, false, false, false>
            <<<dim3(CR / GBM, D / GBN), blk, 0, stream>>>(
            H, WT2T, T2, CR, D, DI, DI, DI, D, 1.0f, nullptr, nullptr);
        context_reduce_kernel<<<dim3(BCHUNK), blk, 0, stream>>>(
            T2, weights, wy, xenc, W_Y, b_Y, C, D, c0, xbuf);
    }

    // ---- predictor block + head ----
    head_kernel<<<dim3(B), blk, 0, stream>>>(
        xbuf, ln_p_g, ln_p_b, W_P1, b_P1, W_P2, b_P2,
        ln_o_g, ln_o_b, W_out, b_out, (float*)d_out);
}
